// Attention_9028021256313
// MI455X (gfx1250) — compile-verified, matrix operand order checked by reading, not run on hardware
//
#include <hip/hip_runtime.h>
#include <hip/hip_bf16.h>
#include <math.h>

// ---------------------------------------------------------------------------
// Attention layer for MI455X (gfx1250, wave32, WMMA + async global->LDS).
// B=8, S=1024, H=2048, NQ=32, NKV=8, D=64.
// Pipeline: cast/transpose -> QKV WMMA GEMM (async-LDS double buffered) ->
//           RoPE -> V-transpose -> flash attention (WMMA QK^T and PV) ->
//           output WMMA GEMM.
// ---------------------------------------------------------------------------

typedef _Float16 v8h  __attribute__((ext_vector_type(8)));
typedef _Float16 v16h __attribute__((ext_vector_type(16)));
typedef float    v8f  __attribute__((ext_vector_type(8)));
typedef int      gv4i __attribute__((vector_size(16)));   // matches builtin sig

#define AS1_ __attribute__((address_space(1)))
#define AS3_ __attribute__((address_space(3)))

#define B_  8
#define S_  1024
#define H_  2048
#define NQ_ 32
#define NKV_ 8
#define D_  64
#define QKVW_ 3072   // 32*64 q + 8*64 k + 8*64 v per token

// ---- CDNA5 async global->LDS support (guarded; fallback = sync copy) ------
#define HAVE_ASYNC_LDS 0
#if defined(__has_builtin)
#if __has_builtin(__builtin_amdgcn_global_load_async_to_lds_b128)
#undef HAVE_ASYNC_LDS
#define HAVE_ASYNC_LDS 1
#endif
#endif

__device__ __forceinline__ void async_copy16(const _Float16* g, _Float16* l) {
#if HAVE_ASYNC_LDS
  __builtin_amdgcn_global_load_async_to_lds_b128(
      (AS1_ gv4i*)g, (AS3_ gv4i*)l, 0, 0);
#else
  *(v8h*)l = *(const v8h*)g;
#endif
}

__device__ __forceinline__ void wait_async() {
#if HAVE_ASYNC_LDS
#if defined(__has_builtin) && __has_builtin(__builtin_amdgcn_s_wait_asynccnt)
  __builtin_amdgcn_s_wait_asynccnt(0);
#else
  asm volatile("s_wait_asynccnt 0" ::: "memory");
#endif
#endif
}

union FragH { v16h v; v8h h[2]; };

// Load a 16x32 f16 fragment (rows = m dim across lanes 0..15, K across regs)
// from row-major memory with leading dimension `ld` (in elements).
// Matches CDNA5 16-bit A-matrix 16x32 layout:
//   lanes 0-15:  regs0-3 K=0..7,  regs4-7 K=16..23
//   lanes 16-31: regs0-3 K=8..15, regs4-7 K=24..31
// B fragments use the identical layout with "row" = output column.
__device__ __forceinline__ v16h load_frag(const _Float16* __restrict__ base, int ld) {
  int lane = threadIdx.x & 31;
  int m    = lane & 15;
  int half = lane >> 4;
  const _Float16* p = base + (size_t)m * ld + 8 * half;
  FragH f;
  f.h[0] = *(const v8h*)(p);        // K = 8*half .. 8*half+7
  f.h[1] = *(const v8h*)(p + 16);   // K = 16+8*half .. 16+8*half+7
  return f.v;
}

__device__ __forceinline__ v8f wmma_f16(v16h a, v16h b, v8f c) {
  return __builtin_amdgcn_wmma_f32_16x16x32_f16(false, a, false, b, (short)0, c,
                                                false, false);
}

// ---------------------------------------------------------------------------
// Elementwise cast f32 -> f16
// ---------------------------------------------------------------------------
__global__ void cast_f32_f16(const float* __restrict__ src,
                             _Float16* __restrict__ dst, size_t n) {
  size_t i = (size_t)blockIdx.x * blockDim.x + threadIdx.x;
  if (i < n) dst[i] = (_Float16)src[i];
}

// ---------------------------------------------------------------------------
// Transpose + cast: src f32 [K x N] row-major -> dst f16 [N x K] row-major.
// ---------------------------------------------------------------------------
__global__ void transpose_cast(const float* __restrict__ src,
                               _Float16* __restrict__ dst, int K, int N) {
  size_t i = (size_t)blockIdx.x * blockDim.x + threadIdx.x;
  size_t total = (size_t)K * N;
  if (i >= total) return;
  int k = (int)(i % K);
  int n = (int)(i / K);
  dst[i] = (_Float16)src[(size_t)k * N + n];
}

// ---------------------------------------------------------------------------
// WMMA GEMM: C[M x N] = A[M x K] * BT[N x K]^T   (A,BT f16; C f32 or f16)
// Block = 256 threads = 8 waves arranged 2x4; wave tile = 64x64 (4x4 frags).
// Block tile = 128 rows x 256 cols, K-step 32, double-buffered LDS staging
// via GLOBAL_LOAD_ASYNC_TO_LDS_B128 (ASYNCcnt). grid = (N/256, M/128).
// LDS: 2 * (128*32 + 256*32) * 2B = 48 KB.
// ---------------------------------------------------------------------------
#define GT_M 128
#define GT_N 256
#define GT_K 32

template <bool HALF_OUT>
__global__ void gemm_wmma(const _Float16* __restrict__ A,
                          const _Float16* __restrict__ BT,
                          void* __restrict__ Cout, int N, int K) {
  __shared__ __align__(16) _Float16 sA[2][GT_M * GT_K];   // 2 x 8 KB
  __shared__ __align__(16) _Float16 sB[2][GT_N * GT_K];   // 2 x 16 KB

  int t    = threadIdx.x;
  int wave = t >> 5;
  int lane = t & 31;
  int n    = lane & 15;
  int half = lane >> 4;
  int row0 = blockIdx.y * GT_M;
  int col0 = blockIdx.x * GT_N;
  int wrow = (wave >> 2) * 64;
  int wcol = (wave & 3) * 64;

  auto stage = [&](int buf, int k0) {
    // A tile: 128x32 halfs = 512 x 16B chunks, 2 per thread
#pragma unroll
    for (int i = 0; i < 2; ++i) {
      int c = t + i * 256;
      int r = c >> 2, cc = (c & 3) * 8;
      async_copy16(A + (size_t)(row0 + r) * K + k0 + cc,
                   &sA[buf][r * GT_K + cc]);
    }
    // B tile: 256x32 halfs = 1024 x 16B chunks, 4 per thread
#pragma unroll
    for (int i = 0; i < 4; ++i) {
      int c = t + i * 256;
      int r = c >> 2, cc = (c & 3) * 8;
      async_copy16(BT + (size_t)(col0 + r) * K + k0 + cc,
                   &sB[buf][r * GT_K + cc]);
    }
  };

  stage(0, 0);
  wait_async();
  __syncthreads();

  const int nk = K / GT_K;
  v8f acc[4][4] = {};
  for (int kt = 0; kt < nk; ++kt) {
    int cur = kt & 1;
    if (kt + 1 < nk) stage(cur ^ 1, (kt + 1) * GT_K);

    v16h a[4], b[4];
#pragma unroll
    for (int i = 0; i < 4; ++i)
      a[i] = load_frag(&sA[cur][(wrow + 16 * i) * GT_K], GT_K);
#pragma unroll
    for (int j = 0; j < 4; ++j)
      b[j] = load_frag(&sB[cur][(wcol + 16 * j) * GT_K], GT_K);
#pragma unroll
    for (int i = 0; i < 4; ++i)
#pragma unroll
      for (int j = 0; j < 4; ++j)
        acc[i][j] = wmma_f16(a[i], b[j], acc[i][j]);

    if (kt + 1 < nk) {
      wait_async();      // next tile fully in LDS (this thread's portion)
      __syncthreads();   // all threads' portions visible; prev reads done
    }
  }

  // C frag layout: lane n = lane&15, reg r -> row = r + 8*(lane>>4).
#pragma unroll
  for (int i = 0; i < 4; ++i)
#pragma unroll
    for (int j = 0; j < 4; ++j)
#pragma unroll
      for (int r = 0; r < 8; ++r) {
        size_t off = (size_t)(row0 + wrow + 16 * i + r + 8 * half) * N +
                     (col0 + wcol + 16 * j + n);
        if (HALF_OUT) ((_Float16*)Cout)[off] = (_Float16)acc[i][j][r];
        else          ((float*)Cout)[off]    = acc[i][j][r];
      }
}

// ---------------------------------------------------------------------------
// RoPE in-place on the f16 QKV buffer [8192 tokens x 3072].
// heads 0..31 = q (col h*64), heads 32..39 = k (col 2048 + (h-32)*64).
// One thread per (token, head, pair d in 0..31).
// ---------------------------------------------------------------------------
__global__ void rope_kernel(_Float16* __restrict__ qkv) {
  const int total = B_ * S_ * (NQ_ + NKV_) * 32;
  int i = blockIdx.x * blockDim.x + threadIdx.x;
  if (i >= total) return;
  int d    = i & 31;
  int rem  = i >> 5;
  int head = rem % (NQ_ + NKV_);
  int tok  = rem / (NQ_ + NKV_);
  int s    = tok & (S_ - 1);
  int col  = (head < NQ_) ? head * D_ : NQ_ * D_ + (head - NQ_) * D_;
  _Float16* p = qkv + (size_t)tok * QKVW_ + col;
  float x1 = (float)p[d];
  float x2 = (float)p[d + 32];
  float freq = __powf(10000.0f, -(float)(2 * d) * (1.0f / 64.0f));
  float sn, cs;
  __sincosf((float)s * freq, &sn, &cs);
  p[d]      = (_Float16)(x1 * cs - x2 * sn);
  p[d + 32] = (_Float16)(x2 * cs + x1 * sn);
}

// ---------------------------------------------------------------------------
// Pack V transposed: Vt[(b*NKV + h)*64 + d][s] = QKV[b*S+s][2560 + h*64 + d]
// ---------------------------------------------------------------------------
__global__ void vt_pack(const _Float16* __restrict__ qkv,
                        _Float16* __restrict__ vt) {
  const int total = B_ * NKV_ * D_ * S_;
  int i = blockIdx.x * blockDim.x + threadIdx.x;
  if (i >= total) return;
  int s   = i & (S_ - 1);
  int rem = i >> 10;
  int d   = rem & (D_ - 1);
  int bh  = rem >> 6;            // b*NKV + h
  int b   = bh >> 3;
  int h   = bh & 7;
  int tok = b * S_ + s;
  vt[(size_t)(bh * D_ + d) * S_ + s] =
      qkv[(size_t)tok * QKVW_ + (NQ_ + NKV_) * D_ + h * D_ + d];
}

// ---------------------------------------------------------------------------
// Flash attention (causal, GQA 4:1). grid = (B*NQ, S/128), 256 threads.
// Each wave owns 16 query rows; loops over 32-key tiles with online softmax.
// QK^T and PV both via v_wmma_f32_16x16x32_f16. P goes C-layout -> LDS ->
// A-layout (per-wave private strip; intra-wave DS ordering covers the RAW).
// ---------------------------------------------------------------------------
__global__ void attn_kernel(const _Float16* __restrict__ qkv,
                            const _Float16* __restrict__ vt,
                            _Float16* __restrict__ aout) {
  __shared__ __align__(16) _Float16 pbuf[8 * 16 * 32];   // 8 KB
  int wave = threadIdx.x >> 5;
  int lane = threadIdx.x & 31;
  int n    = lane & 15;
  int half = lane >> 4;

  int bh  = blockIdx.x;          // b*NQ + h
  int b   = bh >> 5;
  int h   = bh & 31;
  int kvh = h >> 2;              // NQ/NKV = 4
  int qbase = blockIdx.y * 128 + wave * 16;

  const _Float16* Qb = qkv + (size_t)(b * S_ + qbase) * QKVW_ + h * D_;
  v16h a0 = load_frag(Qb, QKVW_);        // d = 0..31
  v16h a1 = load_frag(Qb + 32, QKVW_);   // d = 32..63

  float mrow[8], lrow[8];
#pragma unroll
  for (int r = 0; r < 8; ++r) { mrow[r] = -INFINITY; lrow[r] = 0.0f; }
  v8f acc[4] = {};

  _Float16* myp = pbuf + wave * 512;
  const float scale = 0.125f;            // 1/sqrt(64)
  int ktiles = (qbase + 15) / 32 + 1;

  for (int kt = 0; kt < ktiles; ++kt) {
    int kbase = kt * 32;
    v8f sc[2] = {};
#pragma unroll
    for (int g = 0; g < 2; ++g) {
      const _Float16* Kb =
          qkv + (size_t)(b * S_ + kbase + g * 16) * QKVW_ + NQ_ * D_ + kvh * D_;
      v16h b0 = load_frag(Kb, QKVW_);
      v16h b1 = load_frag(Kb + 32, QKVW_);
      sc[g] = wmma_f16(a0, b0, sc[g]);
      sc[g] = wmma_f16(a1, b1, sc[g]);
    }
    // scale, causal mask, online softmax, write P (f16) into LDS strip
#pragma unroll
    for (int r = 0; r < 8; ++r) {
      int m  = r + 8 * half;
      int qg = qbase + m;
      float sv0 = sc[0][r] * scale;
      float sv1 = sc[1][r] * scale;
      if (kbase + n      > qg) sv0 = -INFINITY;
      if (kbase + 16 + n > qg) sv1 = -INFINITY;
      float mx = fmaxf(sv0, sv1);
#pragma unroll
      for (int off = 1; off < 16; off <<= 1)
        mx = fmaxf(mx, __shfl_xor(mx, off, 32));
      float mnew  = fmaxf(mrow[r], mx);
      float alpha = __expf(mrow[r] - mnew);
      float p0 = __expf(sv0 - mnew);
      float p1 = __expf(sv1 - mnew);
      float ps = p0 + p1;
#pragma unroll
      for (int off = 1; off < 16; off <<= 1)
        ps += __shfl_xor(ps, off, 32);
      lrow[r] = lrow[r] * alpha + ps;
      mrow[r] = mnew;
#pragma unroll
      for (int c = 0; c < 4; ++c) acc[c][r] *= alpha;
      myp[m * 32 + n]      = (_Float16)p0;
      myp[m * 32 + 16 + n] = (_Float16)p1;
    }
    // P as A-fragment (16 rows x 32 keys), then PV accumulation
    v16h pA = load_frag(myp, 32);
#pragma unroll
    for (int c = 0; c < 4; ++c) {
      const _Float16* Vb =
          vt + (size_t)((b * NKV_ + kvh) * D_ + c * 16) * S_ + kbase;
      v16h bv = load_frag(Vb, S_);
      acc[c] = wmma_f16(pA, bv, acc[c]);
    }
  }

  // epilogue: normalize and store f16 rows into [token][2048] at col h*64
#pragma unroll
  for (int r = 0; r < 8; ++r) {
    float inv = 1.0f / lrow[r];
    int qg = qbase + r + 8 * half;
#pragma unroll
    for (int c = 0; c < 4; ++c)
      aout[(size_t)(b * S_ + qg) * (NQ_ * D_) + h * D_ + c * 16 + n] =
          (_Float16)(acc[c][r] * inv);
  }
}

// ---------------------------------------------------------------------------
// Host launcher. Inputs: hidden_states f32, Wq, Wk, Wv, Wo f32.
// Output: f32 [8,1024,2048]. Workspace layout (~147 MB):
//   Xh     f16 [8192 x 2048]   @ 0
//   WqkvT  f16 [3072 x 2048]   @ 33,554,432
//   WoT    f16 [2048 x 2048]   @ 46,137,344
//   QKV    f16 [8192 x 3072]   @ 54,525,952
//   Vt     f16 [4096 x 1024]   @ 104,857,600   (B*NKV*D rows)
//   AOut   f16 [8192 x 2048]   @ 113,246,208
// ---------------------------------------------------------------------------
extern "C" void kernel_launch(void* const* d_in, const int* in_sizes, int n_in,
                              void* d_out, int out_size, void* d_ws,
                              size_t ws_size, hipStream_t stream) {
  const float* hs = (const float*)d_in[0];
  const float* Wq = (const float*)d_in[1];
  const float* Wk = (const float*)d_in[2];
  const float* Wv = (const float*)d_in[3];
  const float* Wo = (const float*)d_in[4];

  char* ws = (char*)d_ws;
  _Float16* Xh    = (_Float16*)(ws);
  _Float16* WqkvT = (_Float16*)(ws + 33554432ull);
  _Float16* WoT   = (_Float16*)(ws + 46137344ull);
  _Float16* QKV   = (_Float16*)(ws + 54525952ull);
  _Float16* Vt    = (_Float16*)(ws + 104857600ull);
  _Float16* AOut  = (_Float16*)(ws + 113246208ull);

  const size_t nX = (size_t)B_ * S_ * H_;          // 16,777,216
  cast_f32_f16<<<(int)((nX + 255) / 256), 256, 0, stream>>>(hs, Xh, nX);

  transpose_cast<<<(H_ * (NQ_ * D_) + 255) / 256, 256, 0, stream>>>(
      Wq, WqkvT, H_, NQ_ * D_);
  transpose_cast<<<(H_ * (NKV_ * D_) + 255) / 256, 256, 0, stream>>>(
      Wk, WqkvT + (size_t)(NQ_ * D_) * H_, H_, NKV_ * D_);
  transpose_cast<<<(H_ * (NKV_ * D_) + 255) / 256, 256, 0, stream>>>(
      Wv, WqkvT + (size_t)((NQ_ + NKV_) * D_) * H_, H_, NKV_ * D_);
  transpose_cast<<<(H_ * H_ + 255) / 256, 256, 0, stream>>>(Wo, WoT, H_, H_);

  // QKV projection: [8192 x 2048] x [2048 x 3072] -> f16 QKV
  gemm_wmma<true><<<dim3(QKVW_ / GT_N, (B_ * S_) / GT_M), 256, 0, stream>>>(
      Xh, WqkvT, (void*)QKV, QKVW_, H_);

  rope_kernel<<<(B_ * S_ * (NQ_ + NKV_) * 32) / 256, 256, 0, stream>>>(QKV);
  vt_pack<<<(B_ * NKV_ * D_ * S_) / 256, 256, 0, stream>>>(QKV, Vt);

  attn_kernel<<<dim3(B_ * NQ_, S_ / 128), 256, 0, stream>>>(QKV, Vt, AOut);

  // Output projection: [8192 x 2048] x [2048 x 2048] -> f32 d_out
  gemm_wmma<false><<<dim3(H_ / GT_N, (B_ * S_) / GT_M), 256, 0, stream>>>(
      AOut, WoT, d_out, H_, H_);
}
